// GraphTransformer_80358838108318
// MI455X (gfx1250) — matrix operand order repaired and checked by reading, not compiled, not run on hardware
//
#include <hip/hip_runtime.h>
#include <hip/hip_bf16.h>
#include <math.h>

typedef __attribute__((ext_vector_type(16))) _Float16 v16h;
typedef __attribute__((ext_vector_type(8)))  _Float16 v8h;
typedef __attribute__((ext_vector_type(8)))  float    v8f;

// ---------------------------------------------------------------- utilities

__global__ void k_zero_u32(unsigned* p, long n) {
    long i  = (long)blockIdx.x * blockDim.x + threadIdx.x;
    long st = (long)gridDim.x * blockDim.x;
    for (; i < n; i += st) p[i] = 0u;
}

// fp32 -> fp16 with nan_to_num(nan=0)
__global__ void k_cvt_f16(const float* __restrict__ x, _Float16* __restrict__ y, long n) {
    long i  = (long)blockIdx.x * blockDim.x + threadIdx.x;
    long st = (long)gridDim.x * blockDim.x;
    for (; i < n; i += st) {
        float v = x[i];
        v = (v != v) ? 0.0f : v;
        y[i] = (_Float16)v;
    }
}

// Pack 4 weight matrices [K][Dmat] (row-major fp32) into a contiguous-k WMMA-B f16
// layout: tiles of 32(K) x 16(N); per tile 512 halves stored as
//   [lane 0..31][elem 0..15], elem i of lane l holds W[k][n] with
//   k = kt*32 + (l<16?0:16) + i , ncat = tn*16 + (l&15)  (ncat = mat*Dmat + n)
// tile order: t = tn * (K/32) + kt
__global__ void k_pack_w(const float* __restrict__ W0, const float* __restrict__ W1,
                         const float* __restrict__ W2, const float* __restrict__ W3,
                         _Float16* __restrict__ Bp, int K, int Dmat) {
    long total = (long)K * Dmat * 4;
    int  ktiles = K >> 5;
    long i  = (long)blockIdx.x * blockDim.x + threadIdx.x;
    long st = (long)gridDim.x * blockDim.x;
    for (; i < total; i += st) {
        int t    = (int)(i >> 9);        // /512 elems per tile
        int r    = (int)(i & 511);
        int lane = r >> 4;
        int el   = r & 15;
        int tn   = t / ktiles;
        int kt   = t - tn * ktiles;
        int k    = kt * 32 + ((lane < 16) ? 0 : 16) + el;
        int ncat = tn * 16 + (lane & 15);
        int mat  = ncat / Dmat;
        int n    = ncat - mat * Dmat;
        const float* W = (mat == 0) ? W0 : (mat == 1) ? W1 : (mat == 2) ? W2 : W3;
        Bp[i] = (_Float16)W[(long)k * Dmat + n];
    }
}

__global__ void k_pack_bias(const float* __restrict__ b0, const float* __restrict__ b1,
                            const float* __restrict__ b2, const float* __restrict__ b3,
                            float* __restrict__ bcat, int Dmat) {
    int i = blockIdx.x * blockDim.x + threadIdx.x;
    if (i < 4 * Dmat) {
        int mat = i / Dmat;
        int n   = i - mat * Dmat;
        const float* b = (mat == 0) ? b0 : (mat == 1) ? b1 : (mat == 2) ? b2 : b3;
        bcat[i] = b[n];
    }
}

// ---------------------------------------------------------------- WMMA GEMM
// Out[N x Ncols] (fp32, +bias) = A[N x K] (f16, row-major) * packed B.
// Block = 4 waves = 16 rows x 256 cols. Each wave holds 4 accumulators
// (4 n-tiles) and reuses one A k-tile across them. The shared A k-tile
// (16x32 halves = 1 KB) is staged into double-buffered LDS by wave 0 using
// CDNA5 async copy (global_load_async_to_lds_b128 + s_wait_asynccnt).
__global__ void __launch_bounds__(128)
k_gemm_f16(const _Float16* __restrict__ A, const _Float16* __restrict__ Bp,
           const float* __restrict__ bias, float* __restrict__ Out,
           int K, int Ncols) {
    __shared__ _Float16 sA[2 * 512];   // two 1 KB buffers, row-major 16 x 64B

    const int lane = threadIdx.x & 31;
    const int wave = threadIdx.x >> 5;
    const int mt   = blockIdx.x;
    const int nt0  = blockIdx.y * 16 + wave * 4;
    const int ktiles = K >> 5;

    // --- async A staging (wave 0 only) -----------------------------------
    // issue j (j=0,1), lane l copies 16B: row = (l>>2)+j*8, chunk = l&3
    const int       arow   = lane >> 2;
    const int       achunk = lane & 3;
    const char*     ag0    = (const char*)(A + (size_t)(mt * 16) * K);
    const unsigned  lbase  = (unsigned)(size_t)(&sA[0]);   // LDS byte offset (addr[31:0])

    auto issueA = [&](int kt) {
        const size_t   rowb = (size_t)K * 2;
        const char*    g    = ag0 + (size_t)kt * 64 + achunk * 16;
        const unsigned lb   = lbase + (unsigned)((kt & 1) * 1024 + achunk * 16);
#pragma unroll
        for (int j = 0; j < 2; ++j) {
            unsigned long long ga = (unsigned long long)(g + (size_t)(arow + j * 8) * rowb);
            unsigned           la = lb + (unsigned)((arow + j * 8) * 64);
            asm volatile("global_load_async_to_lds_b128 %0, %1, off"
                         :: "v"(la), "v"(ga) : "memory");
        }
    };

    if (wave == 0) issueA(0);

    const int m    = lane & 15;
    const int kgrp = lane >> 4;                 // 0: k+{0..15}, 1: k+{16..31}
    const _Float16* s0 = sA + m * 32 + kgrp * 16;  // halves; +8 halves for second run
    const _Float16* bp = Bp + (size_t)nt0 * ktiles * 512 + lane * 16;

    v8f acc[4];
#pragma unroll
    for (int t = 0; t < 4; ++t)
#pragma unroll
        for (int r = 0; r < 8; ++r) acc[t][r] = 0.f;

    for (int kt = 0; kt < ktiles; ++kt) {
        if (wave == 0) asm volatile("s_wait_asynccnt 0x0" ::: "memory");
        __syncthreads();

        const _Float16* sa = s0 + (kt & 1) * 512;
        v8h a0 = *(const v8h*)(sa);
        v8h a1 = *(const v8h*)(sa + 8);
        v16h a;
#pragma unroll
        for (int j = 0; j < 8; ++j) { a[j] = a0[j]; a[j + 8] = a1[j]; }

        const _Float16* bk = bp + (size_t)kt * 512;
#pragma unroll
        for (int t = 0; t < 4; ++t) {
            const _Float16* bt = bk + (size_t)t * ktiles * 512;
            v8h b0 = *(const v8h*)(bt);
            v8h b1 = *(const v8h*)(bt + 8);
            v16h b;
#pragma unroll
            for (int j = 0; j < 8; ++j) { b[j] = b0[j]; b[j + 8] = b1[j]; }
            acc[t] = __builtin_amdgcn_wmma_f32_16x16x32_f16(
                false, a, false, b, (short)0, acc[t], false, false);
        }

        if (wave == 0 && kt + 1 < ktiles) issueA(kt + 1);
        // readers of buffer (kt+1)&1 finished in iteration kt-1, before this
        // iteration's barrier, which precedes the async issue above.
    }

    // C/D layout: elem r of lane l = C[m = r + 8*(l>=16)][n = l&15]
    const int n  = lane & 15;
    const int mr = (lane >> 4) * 8;
#pragma unroll
    for (int t = 0; t < 4; ++t) {
        const int col = (nt0 + t) * 16 + n;
        const float bv = bias[col];
        float* op = Out + (size_t)(mt * 16 + mr) * Ncols + col;
#pragma unroll
        for (int r = 0; r < 8; ++r) { *op = acc[t][r] + bv; op += Ncols; }
    }
}

// ---------------------------------------------------------------- edge stage

__device__ __forceinline__ unsigned f32_enc(float f) {
    unsigned u = __float_as_uint(f);
    return (u & 0x80000000u) ? ~u : (u | 0x80000000u);   // order-preserving
}
__device__ __forceinline__ float f32_dec(unsigned e) {
    unsigned u = (e & 0x80000000u) ? (e & 0x7fffffffu) : ~e;
    return __uint_as_float(u);
}

// one wave per edge: score = scale * dot(q[dst], k[src]); atomicMax into smax
__global__ void k_score(const float* __restrict__ qkvs, const int* __restrict__ ei,
                        float* __restrict__ score, unsigned* __restrict__ smax,
                        int E, int stride, int qoff, int koff, int D, float scale) {
    int gid  = blockIdx.x * blockDim.x + threadIdx.x;
    int e    = gid >> 5;
    int lane = gid & 31;
    if (e >= E) return;
    int src = ei[e];
    int dst = ei[E + e];
    const float* q = qkvs + (size_t)dst * stride + qoff;
    const float* k = qkvs + (size_t)src * stride + koff;
    float s = 0.f;
    for (int c = lane * 4; c < D; c += 128) {
        float4 qv = *(const float4*)(q + c);
        float4 kv = *(const float4*)(k + c);
        s += qv.x * kv.x + qv.y * kv.y + qv.z * kv.z + qv.w * kv.w;
    }
#pragma unroll
    for (int o = 16; o > 0; o >>= 1) s += __shfl_xor(s, o, 32);
    s *= scale;
    if (lane == 0) {
        score[e] = s;
        atomicMax(&smax[dst], f32_enc(s));
    }
}

// thread per edge: ex = exp(score - max[dst]); denom[dst] += ex (score overwritten)
__global__ void k_expsum(float* __restrict__ score, const unsigned* __restrict__ smax,
                         float* __restrict__ denom, const int* __restrict__ ei, int E) {
    int e = blockIdx.x * blockDim.x + threadIdx.x;
    if (e >= E) return;
    int dst  = ei[E + e];
    float mx = f32_dec(smax[dst]);
    float ex = __expf(score[e] - mx);
    score[e] = ex;
    atomicAdd(&denom[dst], ex);
}

// one wave per edge: agg[dst] += ex * v[src]
__global__ void k_agg(const float* __restrict__ qkvs, const float* __restrict__ ex,
                      const int* __restrict__ ei, float* __restrict__ agg,
                      int E, int stride, int voff, int D) {
    int gid  = blockIdx.x * blockDim.x + threadIdx.x;
    int e    = gid >> 5;
    int lane = gid & 31;
    if (e >= E) return;
    int src = ei[e];
    int dst = ei[E + e];
    float w = ex[e];
    const float* v = qkvs + (size_t)src * stride + voff;
    float* a = agg + (size_t)dst * D;
    for (int c = lane * 4; c < D; c += 128) {
        float4 vv = *(const float4*)(v + c);
        atomicAdd(a + c + 0, w * vv.x);
        atomicAdd(a + c + 1, w * vv.y);
        atomicAdd(a + c + 2, w * vv.z);
        atomicAdd(a + c + 3, w * vv.w);
    }
}

// out = agg/denom + skip(=qkvs[...,soff:]); optionally f16 and/or f32 outputs
__global__ void k_finalize(const float* __restrict__ qkvs, const float* __restrict__ agg,
                           const float* __restrict__ denom,
                           _Float16* __restrict__ outH, float* __restrict__ outF,
                           int N, int stride, int soff, int D) {
    long total = (long)N * D;
    long i  = (long)blockIdx.x * blockDim.x + threadIdx.x;
    long st = (long)gridDim.x * blockDim.x;
    for (; i < total; i += st) {
        int n = (int)(i / D);
        int c = (int)(i - (long)n * D);
        float d = denom[n];
        float a = (d > 0.f) ? (agg[i] / d) : 0.f;
        float h = a + qkvs[(size_t)n * stride + soff + c];
        if (outH) outH[i] = (_Float16)h;
        if (outF) outF[i] = h;
    }
}

// one block per channel: avg & max over nodes
__global__ void k_pool(const float* __restrict__ h, float* __restrict__ out, int N, int D) {
    int c = blockIdx.x;
    int t = threadIdx.x;
    float s = 0.f, m = -INFINITY;
    for (int n = t; n < N; n += blockDim.x) {
        float v = h[(size_t)n * D + c];
        s += v;
        m = fmaxf(m, v);
    }
    __shared__ float ss[256];
    __shared__ float sm[256];
    ss[t] = s; sm[t] = m;
    __syncthreads();
    for (int o = 128; o > 0; o >>= 1) {
        if (t < o) { ss[t] += ss[t + o]; sm[t] = fmaxf(sm[t], sm[t + o]); }
        __syncthreads();
    }
    if (t == 0) { out[c] = ss[0] / (float)N; out[D + c] = sm[0]; }
}

// ---------------------------------------------------------------- launcher

extern "C" void kernel_launch(void* const* d_in, const int* in_sizes, int n_in,
                              void* d_out, int out_size, void* d_ws, size_t ws_size,
                              hipStream_t stream) {
    const int DIN = 512, DH = 256, DO = 128;
    const float* x  = (const float*)d_in[0];
    const int*   ei = (const int*)d_in[1];
    // d_in[2] = edge_weight (unused by the reference computation)
    const float *Wq1 = (const float*)d_in[3],  *bq1 = (const float*)d_in[4];
    const float *Wk1 = (const float*)d_in[5],  *bk1 = (const float*)d_in[6];
    const float *Wv1 = (const float*)d_in[7],  *bv1 = (const float*)d_in[8];
    const float *Ws1 = (const float*)d_in[9],  *bs1 = (const float*)d_in[10];
    const float *Wq2 = (const float*)d_in[11], *bq2 = (const float*)d_in[12];
    const float *Wk2 = (const float*)d_in[13], *bk2 = (const float*)d_in[14];
    const float *Wv2 = (const float*)d_in[15], *bv2 = (const float*)d_in[16];
    const float *Ws2 = (const float*)d_in[17], *bs2 = (const float*)d_in[18];

    const int N = in_sizes[0] / DIN;     // 50000 (= 3125 * 16)
    const int E = in_sizes[2];           // 800000

    // ---- workspace layout (256B aligned slabs) ----
    char* ws = (char*)d_ws;
    size_t off = 0;
    auto alloc = [&](size_t bytes) -> char* {
        char* p = ws + off;
        off += (bytes + 255) & ~(size_t)255;
        return p;
    };
    _Float16* xh    = (_Float16*)alloc((size_t)N * DIN * 2);          // 51.2 MB
    float*    qkvs1 = (float*)   alloc((size_t)N * 4 * DH * 4);       // 204.8 MB
    float*    qkvs2 = (float*)   alloc((size_t)N * 4 * DO * 4);       // 102.4 MB
    float*    agg   = (float*)   alloc((size_t)N * DH * 4);           // 51.2 MB (reused L2)
    float*    score = (float*)   alloc((size_t)E * 4);                // 3.2 MB
    unsigned* smax  = (unsigned*)alloc((size_t)N * 4);
    float*    denom = (float*)   alloc((size_t)N * 4);
    _Float16* Wc1   = (_Float16*)alloc((size_t)DIN * 4 * DH * 2);     // 1 MB
    float*    bc1   = (float*)   alloc((size_t)4 * DH * 4);
    _Float16* Wc2   = (_Float16*)alloc((size_t)DH * 4 * DO * 2);
    float*    bc2   = (float*)   alloc((size_t)4 * DO * 4);
    // reuse the (dead after GEMM1) xh slab: f16 hidden h, then fp32 layer-2 output
    _Float16* hh   = (_Float16*)xh;                                   // N*DH*2 = 25.6 MB
    float*    out2 = (float*)((char*)xh + (size_t)N * DH * 2);        // N*DO*4 = 25.6 MB

    const int mtiles = N / 16;   // 3125
    const int ZB = 2048, TB = 256;

    // ================= layer 1 =================
    k_zero_u32<<<ZB, TB, 0, stream>>>(smax, N);
    k_zero_u32<<<ZB, TB, 0, stream>>>((unsigned*)denom, N);
    k_zero_u32<<<ZB, TB, 0, stream>>>((unsigned*)agg, (long)N * DH);

    k_cvt_f16<<<4096, TB, 0, stream>>>(x, xh, (long)N * DIN);
    k_pack_w<<<2048, TB, 0, stream>>>(Wq1, Wk1, Wv1, Ws1, Wc1, DIN, DH);
    k_pack_bias<<<(4 * DH + TB - 1) / TB, TB, 0, stream>>>(bq1, bk1, bv1, bs1, bc1, DH);

    {
        dim3 g(mtiles, (4 * DH) / 256);   // 3125 x 4, block = 16 rows x 256 cols
        k_gemm_f16<<<g, 128, 0, stream>>>(xh, Wc1, bc1, qkvs1, DIN, 4 * DH);
    }

    const int edgeWaveBlocks = (E * 32 + TB - 1) / TB;   // wave per edge
    k_score<<<edgeWaveBlocks, TB, 0, stream>>>(qkvs1, ei, score, smax, E,
                                               4 * DH, 0, DH, DH, 1.0f / sqrtf((float)DH));
    k_expsum<<<(E + TB - 1) / TB, TB, 0, stream>>>(score, smax, denom, ei, E);
    k_agg<<<edgeWaveBlocks, TB, 0, stream>>>(qkvs1, score, ei, agg, E, 4 * DH, 2 * DH, DH);
    k_finalize<<<4096, TB, 0, stream>>>(qkvs1, agg, denom, hh, nullptr, N, 4 * DH, 3 * DH, DH);

    // ================= layer 2 =================
    k_zero_u32<<<ZB, TB, 0, stream>>>(smax, N);
    k_zero_u32<<<ZB, TB, 0, stream>>>((unsigned*)denom, N);
    k_zero_u32<<<ZB, TB, 0, stream>>>((unsigned*)agg, (long)N * DO);

    k_pack_w<<<512, TB, 0, stream>>>(Wq2, Wk2, Wv2, Ws2, Wc2, DH, DO);
    k_pack_bias<<<(4 * DO + TB - 1) / TB, TB, 0, stream>>>(bq2, bk2, bv2, bs2, bc2, DO);

    {
        dim3 g(mtiles, (4 * DO) / 256);   // 3125 x 2
        k_gemm_f16<<<g, 128, 0, stream>>>(hh, Wc2, bc2, qkvs2, DH, 4 * DO);
    }

    k_score<<<edgeWaveBlocks, TB, 0, stream>>>(qkvs2, ei, score, smax, E,
                                               4 * DO, 0, DO, DO, 1.0f / sqrtf((float)DO));
    k_expsum<<<(E + TB - 1) / TB, TB, 0, stream>>>(score, smax, denom, ei, E);
    k_agg<<<edgeWaveBlocks, TB, 0, stream>>>(qkvs2, score, ei, agg, E, 4 * DO, 2 * DO, DO);
    k_finalize<<<4096, TB, 0, stream>>>(qkvs2, agg, denom, nullptr, out2, N, 4 * DO, 3 * DO, DO);

    // ================= pooling =================
    k_pool<<<DO, TB, 0, stream>>>(out2, (float*)d_out, N, DO);
}
